// TransformerBlock_69226282877353
// MI455X (gfx1250) — matrix operand order repaired and checked by reading, not compiled, not run on hardware
//
#include <hip/hip_runtime.h>
#include <hip/hip_bf16.h>

// ---------------------------------------------------------------------------
// Problem constants (match reference): D=1024 H=16 FF=4096 E=2 K=2 B=2 S=2048
// ---------------------------------------------------------------------------
constexpr int Dm  = 1024;
constexpr int Hh  = 16;
constexpr int FFd = 4096;
constexpr int Ee  = 2;
constexpr int Bb  = 2;
constexpr int Ss  = 2048;
constexpr int Tt  = Bb * Ss;       // 4096 tokens
constexpr int HD  = Dm / Hh;       // 64 head dim

typedef __attribute__((ext_vector_type(16))) __bf16 v16bf;
typedef __attribute__((ext_vector_type(8)))  __bf16 v8bf;   // 16 bytes
typedef __attribute__((ext_vector_type(8)))  float  v8f;

static __device__ __forceinline__ v8f wmma_bf16(v16bf a, v16bf b, v8f c) {
  // D = A(16x32 bf16) * B(32x16 bf16) + C(16x16 f32)
  return __builtin_amdgcn_wmma_f32_16x16x32_bf16(
      /*neg_a=*/false, a, /*neg_b=*/false, b,
      /*c_mod=*/(short)0, c, /*reuse_a=*/false, /*reuse_b=*/false);
}

// ---------------------------------------------------------------------------
// f32 -> bf16 elementwise conversion (grid-stride)
// ---------------------------------------------------------------------------
__global__ __launch_bounds__(256) void cvt_bf16_kernel(
    const float* __restrict__ in, __bf16* __restrict__ out, size_t n) {
  size_t i = (size_t)blockIdx.x * blockDim.x + threadIdx.x;
  size_t stride = (size_t)gridDim.x * blockDim.x;
  for (; i < n; i += stride) out[i] = (__bf16)in[i];
}

// ---------------------------------------------------------------------------
// LayerNorm: one block per token, D=1024, writes optional f32 and bf16 copies
// ---------------------------------------------------------------------------
__global__ __launch_bounds__(256) void ln_kernel(
    const float* __restrict__ x, const float* __restrict__ g,
    const float* __restrict__ beta,
    float* __restrict__ outF, __bf16* __restrict__ outB) {
  __shared__ float s1[256], s2[256];
  const int t = blockIdx.x;
  const float* xr = x + (size_t)t * Dm;
  float a0 = 0.f, a1 = 0.f;
  for (int i = threadIdx.x; i < Dm; i += 256) {
    float v = xr[i]; a0 += v; a1 += v * v;
  }
  s1[threadIdx.x] = a0; s2[threadIdx.x] = a1;
  __syncthreads();
  for (int off = 128; off > 0; off >>= 1) {
    if (threadIdx.x < off) {
      s1[threadIdx.x] += s1[threadIdx.x + off];
      s2[threadIdx.x] += s2[threadIdx.x + off];
    }
    __syncthreads();
  }
  const float mu   = s1[0] * (1.0f / Dm);
  const float var  = s2[0] * (1.0f / Dm) - mu * mu;
  const float rstd = rsqrtf(var + 1e-5f);
  for (int i = threadIdx.x; i < Dm; i += 256) {
    float v = (xr[i] - mu) * rstd * g[i] + beta[i];
    if (outF) outF[(size_t)t * Dm + i] = v;
    if (outB) outB[(size_t)t * Dm + i] = (__bf16)v;
  }
}

// ---------------------------------------------------------------------------
// Tiled WMMA GEMM:  C[M,N] = act(A[M,K] * W[N,K]^T + bias) (+residual)
// 256 thr (8 wave32) per 128x128 C tile; waves arranged 4x2, each wave owns a
// 32x64 sub-tile = 2x4 WMMA tiles -> 8 independent accumulator chains and
// 8 WMMAs per 12 ds_load_b128 per K-step (K stepped by 32, double-buffered).
// LDS tiles are stored in WMMA *fragment order*: each lane's 16 bf16 live in
// one contiguous 32B slot, so a fragment load is 2x ds_load_b128 and staging
// is one global_load_b128 + one ds_store_b128 per 8-element run.
//
// ISA A 16x32 bf16 layout:  lane = mloc + 16*khalf,
//   j=0..7  <-> k = khalf*8 + j          (contiguous run of 8)
//   j=8..15 <-> k = 16 + khalf*8 + (j-8) (contiguous run of 8)
// ISA B 32x16 bf16 layout:  lane = nloc + 16*khalf, same k order as A.
// ---------------------------------------------------------------------------
__global__ __launch_bounds__(256) void gemm_bf16_kernel(
    const __bf16* __restrict__ A, const __bf16* __restrict__ W,
    const float* __restrict__ bias, const float* __restrict__ residual,
    float* __restrict__ outF, __bf16* __restrict__ outB,
    int M, int N, int K, int gelu) {
  __shared__ __align__(32) __bf16 Afrag[2][8][32][16];  // [buf][tm][lane][j]
  __shared__ __align__(32) __bf16 Bfrag[2][8][32][16];  // [buf][tn][lane][j]

  const int tid   = threadIdx.x;
  const int wave  = tid >> 5;
  const int lane  = tid & 31;
  const int mloc  = lane & 15;
  const int khalf = lane >> 4;
  const int wr    = wave >> 1;     // 0..3: tile rows wr*2 + {0,1}
  const int wc    = wave & 1;      // 0..1: tile cols wc*4 + {0..3}
  const int m0    = blockIdx.y * 128;
  const int n0    = blockIdx.x * 128;

  // stage one K-step: 128x32 A and 32x128 B, both in fragment order.
  // 512 8-element runs per matrix; each thread moves 2 runs per matrix.
  auto stage = [&](int buf, int k0) {
#pragma unroll
    for (int u = tid; u < 512; u += 256) {
      const int jh = u & 1;
      const int ln = (u >> 1) & 31;
      const int tl = u >> 6;                     // tile index 0..7
      const int row  = tl * 16 + (ln & 15);
      const int ka   = jh * 16 + (ln >> 4) * 8;  // A-layout k offset
      const int kb   = jh * 16 + (ln >> 4) * 8;  // B k offset (same order as A)
      *(v8bf*)&Afrag[buf][tl][ln][jh * 8] =
          *(const v8bf*)&A[(size_t)(m0 + row) * K + k0 + ka];
      *(v8bf*)&Bfrag[buf][tl][ln][jh * 8] =
          *(const v8bf*)&W[(size_t)(n0 + row) * K + k0 + kb];
    }
  };

  v8f acc[2][4];
#pragma unroll
  for (int tr = 0; tr < 2; ++tr)
#pragma unroll
    for (int tc = 0; tc < 4; ++tc)
#pragma unroll
      for (int r = 0; r < 8; ++r) acc[tr][tc][r] = 0.0f;

  const int nk = K >> 5;
  stage(0, 0);
  for (int kt = 0; kt < nk; ++kt) {
    __syncthreads();
    const int cur = kt & 1;
    if (kt + 1 < nk) {
      stage(cur ^ 1, (kt + 1) * 32);
      // L2 prefetch two steps ahead (global_prefetch_b8); clamp, no branch
      const int kpf = (kt + 2 < nk) ? (kt + 2) * 32 : kt * 32;
      __builtin_prefetch(&A[(size_t)(m0 + (tid & 127)) * K + kpf], 0, 3);
      __builtin_prefetch(&W[(size_t)(n0 + (tid & 127)) * K + kpf], 0, 3);
    }
    v16bf af[2];
#pragma unroll
    for (int tr = 0; tr < 2; ++tr)
      af[tr] = *(const v16bf*)&Afrag[cur][wr * 2 + tr][lane][0];
    v16bf bf[4];
#pragma unroll
    for (int tc = 0; tc < 4; ++tc)
      bf[tc] = *(const v16bf*)&Bfrag[cur][wc * 4 + tc][lane][0];
#pragma unroll
    for (int tr = 0; tr < 2; ++tr)
#pragma unroll
      for (int tc = 0; tc < 4; ++tc)
        acc[tr][tc] = wmma_bf16(af[tr], bf[tc], acc[tr][tc]);
  }

  // epilogue: bias (+GELU) (+residual), write f32 and/or bf16
#pragma unroll
  for (int tr = 0; tr < 2; ++tr) {
#pragma unroll
    for (int tc = 0; tc < 4; ++tc) {
      const int col = n0 + (wc * 4 + tc) * 16 + mloc;
      const float bv = bias ? bias[col] : 0.0f;
#pragma unroll
      for (int r = 0; r < 8; ++r) {
        const int row = m0 + (wr * 2 + tr) * 16 + khalf * 8 + r;  // C/D layout
        float v = acc[tr][tc][r] + bv;
        if (gelu) v = 0.5f * v * (1.0f + erff(v * 0.70710678118654752f));
        if (residual) v += residual[(size_t)row * N + col];
        const size_t idx = (size_t)row * N + col;
        if (outF) outF[idx] = v;
        if (outB) outB[idx] = (__bf16)v;
      }
    }
  }
}

// ---------------------------------------------------------------------------
// Flash attention: one block per (b, h, 64-query tile). K/V streamed in
// 64-key chunks. Q fragments staged once (fragment-order LDS) and hoisted to
// registers; K fragments staged in B-layout (all-vector); P written by the
// softmax pass directly in A-fragment layout. Online softmax in f32.
// ---------------------------------------------------------------------------
__global__ __launch_bounds__(256) void attention_kernel(
    const __bf16* __restrict__ qkv, __bf16* __restrict__ o) {
  __shared__ __align__(32) __bf16 Qfrag[4][2][32][16];  // [tm][kb][lane][j]
  __shared__ __align__(32) __bf16 Kfrag[4][2][32][16];  // [tn][kb][lane][j]
  __shared__ __align__(32) __bf16 Pfrag[4][2][32][16];  // [tm][kb][lane][j]
  __shared__ __align__(32) __bf16 Vs[64][72];           // [key][d] (+pad)
  __shared__ float  Sc[64][65];
  __shared__ float  rowm[64], rowl[64], rowscale[64];

  const int tid   = threadIdx.x;
  const int wave  = tid >> 5;
  const int lane  = tid & 31;
  const int mloc  = lane & 15;
  const int khalf = lane >> 4;
  const int tm    = wave >> 1;

  const int blk = blockIdx.x;
  const int qb  = blk & 31;          // S/64 query tiles
  const int h   = (blk >> 5) & 15;
  const int b   = blk >> 9;
  const int t0  = b * Ss + qb * 64;

  // ---- stage Q once, fragment order, pre-scaled by 1/sqrt(hd) ----
  for (int unit = tid; unit < 512; unit += 256) {
    const int jh   = unit & 1;
    const int ln   = (unit >> 1) & 31;
    const int kb   = (unit >> 6) & 1;
    const int tms  = unit >> 7;
    const int m    = tms * 16 + (ln & 15);
    const int d0   = kb * 32 + jh * 16 + (ln >> 4) * 8;
    v8bf q8 = *(const v8bf*)&qkv[(size_t)(t0 + m) * (3 * Dm) + h * HD + d0];
    v8bf qs;
#pragma unroll
    for (int e = 0; e < 8; ++e) qs[e] = (__bf16)((float)q8[e] * 0.125f);
    *(v8bf*)&Qfrag[tms][kb][ln][jh * 8] = qs;
  }
  if (tid < 64) { rowm[tid] = -3.0e38f; rowl[tid] = 0.0f; }
  __syncthreads();

  // Q fragments are chunk-invariant: hoist into registers
  v16bf qf[2];
#pragma unroll
  for (int kb = 0; kb < 2; ++kb)
    qf[kb] = *(const v16bf*)&Qfrag[tm][kb][lane][0];

  v8f acc[2];
#pragma unroll
  for (int i = 0; i < 2; ++i)
#pragma unroll
    for (int r = 0; r < 8; ++r) acc[i][r] = 0.0f;

  for (int jc = 0; jc < Ss / 64; ++jc) {
    const int kt0 = b * Ss + jc * 64;

    // ---- stage K fragments (B-layout, vector copies) + V tile ----
    for (int unit = tid; unit < 512; unit += 256) {
      const int jh  = unit & 1;
      const int ln  = (unit >> 1) & 31;
      const int kb  = (unit >> 6) & 1;
      const int tns = unit >> 7;
      const int key = tns * 16 + (ln & 15);
      const int d0  = kb * 32 + jh * 16 + (ln >> 4) * 8;
      *(v8bf*)&Kfrag[tns][kb][ln][jh * 8] =
          *(const v8bf*)&qkv[(size_t)(kt0 + key) * (3 * Dm) + Dm + h * HD + d0];
    }
    for (int unit = tid; unit < 512; unit += 256) {
      const int row = unit >> 3;
      const int c8  = (unit & 7) << 3;
      *(v8bf*)&Vs[row][c8] =
          *(const v8bf*)&qkv[(size_t)(kt0 + row) * (3 * Dm) + 2 * Dm + h * HD + c8];
    }
    __syncthreads();

    // ---- S_blk = Q * K^T (2 WMMAs per 16x16 tile) ----
#pragma unroll
    for (int i = 0; i < 2; ++i) {
      const int tn = (wave & 1) * 2 + i;
      v8f s;
#pragma unroll
      for (int r = 0; r < 8; ++r) s[r] = 0.0f;
#pragma unroll
      for (int kb = 0; kb < 2; ++kb) {
        const v16bf bf = *(const v16bf*)&Kfrag[tn][kb][lane][0];
        s = wmma_bf16(qf[kb], bf, s);
      }
#pragma unroll
      for (int r = 0; r < 8; ++r)
        Sc[tm * 16 + khalf * 8 + r][tn * 16 + mloc] = s[r];
    }
    __syncthreads();

    // ---- online softmax; write P directly in A-fragment layout ----
    if (tid < 64) {
      const int row = tid;
      const int ptm = row >> 4, pml = row & 15;
      float mc = -3.0e38f;
      for (int c = 0; c < 64; ++c) mc = fmaxf(mc, Sc[row][c]);
      const float mold = rowm[row];
      const float mnew = fmaxf(mold, mc);
      const float scl  = __expf(mold - mnew);
      float ls = 0.f;
      for (int c = 0; c < 64; ++c) {
        const float p = __expf(Sc[row][c] - mnew);
        const int kb = c >> 5, kl = c & 31;
        const int kh = (kl >> 3) & 1;
        const int j  = (kl & 7) | ((kl >> 4) << 3);
        Pfrag[ptm][kb][pml + 16 * kh][j] = (__bf16)p;
        ls += p;
      }
      rowm[row]     = mnew;
      rowl[row]     = rowl[row] * scl + ls;
      rowscale[row] = scl;
    }
    __syncthreads();

    // ---- O = O*scale + P * V ----
#pragma unroll
    for (int i = 0; i < 2; ++i) {
      const int tn = (wave & 1) * 2 + i;
#pragma unroll
      for (int r = 0; r < 8; ++r)
        acc[i][r] *= rowscale[tm * 16 + khalf * 8 + r];
#pragma unroll
      for (int kb = 0; kb < 2; ++kb) {
        const v16bf af = *(const v16bf*)&Pfrag[tm][kb][lane][0];
        v16bf bf;  // V gather: B element (k=key_local, n=d)
#pragma unroll
        for (int j = 0; j < 16; ++j)
          bf[j] = Vs[kb * 32 + (j >> 3) * 16 + khalf * 8 + (j & 7)][tn * 16 + mloc];
        acc[i] = wmma_bf16(af, bf, acc[i]);
      }
    }
    __syncthreads();
  }

  // ---- normalize and write o in [T][D] layout (col = h*64 + d) ----
#pragma unroll
  for (int i = 0; i < 2; ++i) {
    const int tn  = (wave & 1) * 2 + i;
    const int col = h * HD + tn * 16 + mloc;
#pragma unroll
    for (int r = 0; r < 8; ++r) {
      const int row = tm * 16 + khalf * 8 + r;
      const float v = acc[i][r] / rowl[row];
      o[(size_t)(t0 + row) * Dm + col] = (__bf16)v;
    }
  }
}

// ---------------------------------------------------------------------------
// Gate: logits -> softmax. With E=2,K=2 the renormalized top-k combine weights
// equal the softmax probs exactly.
// ---------------------------------------------------------------------------
__global__ __launch_bounds__(128) void gate_kernel(
    const __bf16* __restrict__ n2, const float* __restrict__ gw,
    const float* __restrict__ gb, float* __restrict__ probs,
    float* __restrict__ cw) {
  __shared__ float sh0[128], sh1[128];
  const int t = blockIdx.x;
  float d0 = 0.f, d1 = 0.f;
  for (int i = threadIdx.x; i < Dm; i += 128) {
    float v = (float)n2[(size_t)t * Dm + i];
    d0 += v * gw[i];
    d1 += v * gw[Dm + i];
  }
  sh0[threadIdx.x] = d0; sh1[threadIdx.x] = d1;
  __syncthreads();
  for (int off = 64; off > 0; off >>= 1) {
    if (threadIdx.x < off) {
      sh0[threadIdx.x] += sh0[threadIdx.x + off];
      sh1[threadIdx.x] += sh1[threadIdx.x + off];
    }
    __syncthreads();
  }
  if (threadIdx.x == 0) {
    float l0 = sh0[0] + gb[0], l1 = sh1[0] + gb[1];
    float m  = fmaxf(l0, l1);
    float e0 = __expf(l0 - m), e1 = __expf(l1 - m);
    float inv = 1.0f / (e0 + e1);
    float p0 = e0 * inv, p1 = e1 * inv;
    probs[t * 2 + 0] = p0; probs[t * 2 + 1] = p1;
    cw[t * 2 + 0] = p0;    cw[t * 2 + 1] = p1;
  }
}

// Deterministic Switch load-balance loss (single block tree reduce, no float
// atomics -> bitwise reproducible across graph replays).
__global__ __launch_bounds__(256) void loss_kernel(
    const float* __restrict__ probs, float* __restrict__ out_loss) {
  __shared__ float f0s[256], f1s[256], p0s[256], p1s[256];
  float f0 = 0.f, f1 = 0.f, p0 = 0.f, p1 = 0.f;
  for (int t = threadIdx.x; t < Tt; t += 256) {
    float a = probs[t * 2], b = probs[t * 2 + 1];
    p0 += a; p1 += b;
    if (a >= b) f0 += 1.0f; else f1 += 1.0f;   // argmax, first-index tiebreak
  }
  f0s[threadIdx.x] = f0; f1s[threadIdx.x] = f1;
  p0s[threadIdx.x] = p0; p1s[threadIdx.x] = p1;
  __syncthreads();
  for (int off = 128; off > 0; off >>= 1) {
    if (threadIdx.x < off) {
      f0s[threadIdx.x] += f0s[threadIdx.x + off];
      f1s[threadIdx.x] += f1s[threadIdx.x + off];
      p0s[threadIdx.x] += p0s[threadIdx.x + off];
      p1s[threadIdx.x] += p1s[threadIdx.x + off];
    }
    __syncthreads();
  }
  if (threadIdx.x == 0) {
    const float invT = 1.0f / (float)Tt;
    float loss = 0.01f * (float)Ee *
        ((f0s[0] * invT) * (p0s[0] * invT) + (f1s[0] * invT) * (p1s[0] * invT));
    out_loss[0] = loss;
  }
}

// out = x1 + sum_e cw[t,e] * eo[e,t,:]
__global__ __launch_bounds__(256) void combine_kernel(
    const float* __restrict__ x1, const float* __restrict__ cw,
    const float* __restrict__ eo, float* __restrict__ out) {
  size_t idx = (size_t)blockIdx.x * 256 + threadIdx.x;   // T*D elements
  size_t t = idx / Dm;
  out[idx] = x1[idx] + cw[t * 2 + 0] * eo[idx]
                     + cw[t * 2 + 1] * eo[(size_t)Tt * Dm + idx];
}

// ---------------------------------------------------------------------------
// Host orchestration
// ---------------------------------------------------------------------------
extern "C" void kernel_launch(void* const* d_in, const int* in_sizes, int n_in,
                              void* d_out, int out_size, void* d_ws, size_t ws_size,
                              hipStream_t stream) {
  const float* x    = (const float*)d_in[0];
  const float* inw  = (const float*)d_in[1];
  const float* inb  = (const float*)d_in[2];
  const float* outw = (const float*)d_in[3];
  const float* outb = (const float*)d_in[4];
  const float* ln1g = (const float*)d_in[5];
  const float* ln1b = (const float*)d_in[6];
  const float* ln2g = (const float*)d_in[7];
  const float* ln2b = (const float*)d_in[8];
  const float* gw   = (const float*)d_in[9];
  const float* gb   = (const float*)d_in[10];
  const float* w1   = (const float*)d_in[11];
  const float* b1   = (const float*)d_in[12];
  const float* w2   = (const float*)d_in[13];
  const float* b2   = (const float*)d_in[14];

  // workspace carve-out (~205 MB)
  char* ws = (char*)d_ws;
  size_t off = 0;
  auto carve = [&](size_t bytes) {
    size_t o = off;
    off += (bytes + 255) & ~(size_t)255;
    return (void*)(ws + o);
  };
  __bf16* normed_bf = (__bf16*)carve((size_t)Tt * Dm * 2);
  __bf16* qkv_bf    = (__bf16*)carve((size_t)Tt * 3 * Dm * 2);
  __bf16* o_bf      = (__bf16*)carve((size_t)Tt * Dm * 2);
  float*  x1        = (float*) carve((size_t)Tt * Dm * 4);
  __bf16* n2_bf     = (__bf16*)carve((size_t)Tt * Dm * 2);
  __bf16* h_bf      = (__bf16*)carve((size_t)Ee * Tt * FFd * 2);
  float*  eo        = (float*) carve((size_t)Ee * Tt * Dm * 4);
  float*  cw        = (float*) carve((size_t)Tt * Ee * 4);
  __bf16* inw_bf    = (__bf16*)carve((size_t)3 * Dm * Dm * 2);
  __bf16* outw_bf   = (__bf16*)carve((size_t)Dm * Dm * 2);
  __bf16* w1_bf     = (__bf16*)carve((size_t)Ee * FFd * Dm * 2);
  __bf16* w2_bf     = (__bf16*)carve((size_t)Ee * Dm * FFd * 2);

  // output sections: out | load_loss | probs | normed (present_kv[0])
  float* out        = (float*)d_out;
  float* out_loss   = out + (size_t)Tt * Dm;
  float* out_probs  = out_loss + 1;
  float* out_normed = out_probs + (size_t)Tt * Ee;

  // 1) weight precision conversion f32 -> bf16
  cvt_bf16_kernel<<<2048, 256, 0, stream>>>(inw,  inw_bf,  (size_t)3 * Dm * Dm);
  cvt_bf16_kernel<<<2048, 256, 0, stream>>>(outw, outw_bf, (size_t)Dm * Dm);
  cvt_bf16_kernel<<<4096, 256, 0, stream>>>(w1,   w1_bf,   (size_t)Ee * FFd * Dm);
  cvt_bf16_kernel<<<4096, 256, 0, stream>>>(w2,   w2_bf,   (size_t)Ee * Dm * FFd);

  // 2) LN1 -> normed (f32 into present_kv output) + bf16 activation
  ln_kernel<<<Tt, 256, 0, stream>>>(x, ln1g, ln1b, out_normed, normed_bf);

  // 3) QKV projection: [T,1024] x [3072,1024]^T -> bf16 qkv
  gemm_bf16_kernel<<<dim3(3 * Dm / 128, Tt / 128), 256, 0, stream>>>(
      normed_bf, inw_bf, inb, nullptr, nullptr, qkv_bf, Tt, 3 * Dm, Dm, 0);

  // 4) flash attention: B*H*(S/64) = 1024 blocks
  attention_kernel<<<Bb * Hh * (Ss / 64), 256, 0, stream>>>(qkv_bf, o_bf);

  // 5) out-proj + residual x -> x1 (f32)
  gemm_bf16_kernel<<<dim3(Dm / 128, Tt / 128), 256, 0, stream>>>(
      o_bf, outw_bf, outb, x, x1, nullptr, Tt, Dm, Dm, 0);

  // 6) LN2 -> bf16
  ln_kernel<<<Tt, 256, 0, stream>>>(x1, ln2g, ln2b, nullptr, n2_bf);

  // 7) gate softmax -> probs (output) + combine weights
  gate_kernel<<<Tt, 128, 0, stream>>>(n2_bf, gw, gb, out_probs, cw);

  // 8) dense per-expert FFN: h = gelu(n2 @ w1^T + b1) ; eo = h @ w2^T + b2
  for (int e = 0; e < Ee; ++e) {
    gemm_bf16_kernel<<<dim3(FFd / 128, Tt / 128), 256, 0, stream>>>(
        n2_bf, w1_bf + (size_t)e * FFd * Dm, b1 + (size_t)e * FFd,
        nullptr, nullptr, h_bf + (size_t)e * Tt * FFd, Tt, FFd, Dm, 1);
    gemm_bf16_kernel<<<dim3(Dm / 128, Tt / 128), 256, 0, stream>>>(
        h_bf + (size_t)e * Tt * FFd, w2_bf + (size_t)e * Dm * FFd,
        b2 + (size_t)e * Dm, nullptr, eo + (size_t)e * Tt * Dm, nullptr,
        Tt, Dm, FFd, 0);
  }

  // 9) weighted combine + residual -> out
  combine_kernel<<<(Tt * Dm) / 256, 256, 0, stream>>>(x1, cw, eo, out);

  // 10) deterministic load-balance loss
  loss_kernel<<<1, 256, 0, stream>>>(out_probs, out_loss);
}
